// LSTM_77592879170160
// MI455X (gfx1250) — matrix operand order repaired and checked by reading, not compiled, not run on hardware
//
#include <hip/hip_runtime.h>
#include <hip/hip_bf16.h>

typedef __attribute__((ext_vector_type(16))) __bf16 v16bf;
typedef __attribute__((ext_vector_type(8)))  __bf16 bf16x8;
typedef __attribute__((ext_vector_type(8)))  float  v8f;
typedef int v4i_gcc __attribute__((vector_size(16)));   // matches builtin param type

#define AS1 __attribute__((address_space(1)))
#define AS3 __attribute__((address_space(3)))

#define BATCH 64
#define SEQ   2048
#define INF_  256
#define HID_  256
#define NCOMB 1024              // 4 gates * HID
#define MROWS (BATCH*SEQ)       // 131072
#define NWG_LSTM 16

#if __has_builtin(__builtin_amdgcn_global_load_async_to_lds_b128)
#define HAVE_ASYNC_LDS 1
#else
#define HAVE_ASYNC_LDS 0
#endif

union Frag16 { __bf16 e[16]; bf16x8 v8[2]; };

__device__ __forceinline__ float sigmoidf_(float x){ return 1.0f/(1.0f+__expf(-x)); }
__device__ __forceinline__ float tanhf_(float x){ return 1.0f - 2.0f/(__expf(2.0f*x)+1.0f); }

__device__ __forceinline__ v16bf make_frag(bf16x8 lo, bf16x8 hi){
  v16bf r;
#pragma unroll
  for (int i=0;i<8;++i){ r[i]=lo[i]; r[i+8]=hi[i]; }
  return r;
}

// ---------------------------------------------------------------- init state
__global__ void k_init(__bf16* hbuf, unsigned* bar){
  int tid = threadIdx.x;
  for (int i = tid; i < 2*BATCH*HID_; i += blockDim.x) hbuf[i] = (__bf16)0.0f;
  if (tid < 4) bar[tid] = 0u;
}

// ---------------------------------------------------------------- bias merge
__global__ void k_bias(const float* bii,const float* bhi,const float* bif,const float* bhf,
                       const float* big,const float* bhg,const float* bio,const float* bho,
                       float* bias){
  int c = blockIdx.x*blockDim.x + threadIdx.x;     // 0..1023
  int g = c >> 8, j = c & 255;
  const float* bi[4] = {bii,bif,big,bio};
  const float* bh[4] = {bhi,bhf,bhg,bho};
  bias[c] = bi[g][j] + bh[g][j];
}

// ------------------------------------------------ weights -> bf16 WMMA frags
// Fragment-major storage: tile T = kt*64+nt holds 32 lanes x 16 bf16 contiguous.
// Element (lane,v) = W[kt*32 + (lane>>4)*16 + v][nt*16 + (lane&15)]
__global__ void k_wprep(const float* Wii,const float* Whi,const float* Wif,const float* Whf,
                        const float* Wig,const float* Whg,const float* Wio,const float* Who,
                        __bf16* wfi, __bf16* wfh){
  int tid  = blockIdx.x*blockDim.x + threadIdx.x;  // 0..32767
  int set  = tid >> 14;                            // 0:input weights 1:hidden weights
  int rem  = tid & 16383;
  int tile = rem >> 5;                             // 0..511
  int lane = rem & 31;
  int kt = tile >> 6;
  int nt = tile & 63;
  int col = nt*16 + (lane & 15);
  int g = col >> 8, j = col & 255;
  const float* srcI[4] = {Wii,Wif,Wig,Wio};
  const float* srcH[4] = {Whi,Whf,Whg,Who};
  const float* src = (set==0) ? srcI[g] : srcH[g];
  int k0 = kt*32 + (lane>>4)*16;
  Frag16 tmp;
#pragma unroll
  for (int v=0; v<16; ++v) tmp.e[v] = (__bf16)src[(size_t)(k0+v)*HID_ + j];
  __bf16* dst = ((set==0)?wfi:wfh) + ((size_t)tile*32 + lane)*16;
  *(bf16x8*)dst       = tmp.v8[0];
  *(bf16x8*)(dst + 8) = tmp.v8[1];
}

// ---------------------------------------------------------------- x -> bf16
__global__ void k_xconv(const float* __restrict__ x, __bf16* __restrict__ xbf){
  size_t i = ((size_t)blockIdx.x*blockDim.x + threadIdx.x)*4;
  float4 f = *(const float4*)(x + i);
  union { __bf16 e[4]; unsigned long long u; } t;
  t.e[0]=(__bf16)f.x; t.e[1]=(__bf16)f.y; t.e[2]=(__bf16)f.z; t.e[3]=(__bf16)f.w;
  *(unsigned long long*)(xbf + i) = t.u;
}

// -------------------------------------------- phase 1: P = x@W_i + (b_i+b_h)
// One 16x16 output tile per wave, K-loop of 8 bf16 WMMAs.
__global__ void __launch_bounds__(256) k_gemm(const __bf16* __restrict__ xbf,
                                              const __bf16* __restrict__ wfi,
                                              const float*  __restrict__ bias,
                                              float* __restrict__ P){
  int wave = threadIdx.x >> 5;
  int lane = threadIdx.x & 31;
  int half = lane >> 4;
  int nlo  = lane & 15;
  int tile = blockIdx.x*8 + wave;      // 0..524287
  int mt = tile >> 6;
  int nt = tile & 63;
  int arow = mt*16 + nlo;
  int ccol = nt*16 + nlo;
  v8f acc;
  float bv = bias[ccol];
#pragma unroll
  for (int r=0;r<8;++r) acc[r]=bv;
  const __bf16* Arow = xbf + (size_t)arow*INF_;
#pragma unroll
  for (int kt=0; kt<8; ++kt){
    int kc = kt*32 + half*16;
    bf16x8 a0 = *(const bf16x8*)(Arow + kc);
    bf16x8 a1 = *(const bf16x8*)(Arow + kc + 8);
    const __bf16* bp = wfi + ((size_t)(kt*64 + nt)*32 + lane)*16;
    bf16x8 b0 = *(const bf16x8*)bp;
    bf16x8 b1 = *(const bf16x8*)(bp + 8);
    acc = __builtin_amdgcn_wmma_f32_16x16x32_bf16(false, make_frag(a0,a1),
                                                  false, make_frag(b0,b1),
                                                  (short)0, acc, false, false);
  }
  int rb_base = mt*16 + half*8;
#pragma unroll
  for (int r=0;r<8;++r){
    int rb = rb_base + r;            // row in [BATCH*SEQ): rb = b*SEQ + t
    int b = rb >> 11;
    int t = rb & 2047;
    P[((size_t)t*BATCH + b)*NCOMB + ccol] = acc[r];   // P laid out [t][b][1024]
  }
}

// ----------------------- phase 2: persistent recurrent scan, 16 WGs x 16 waves
// WG wg owns hidden cols [wg*16, wg*16+16) of all 4 gates; W_h slice lives in LDS.
// h[t] is staged into LDS each step via the CDNA5 async Global->LDS path.
__global__ void __launch_bounds__(512) k_lstm(const __bf16* __restrict__ wfh,
                                              const float* __restrict__ P,
                                              __bf16* __restrict__ hbuf,
                                              float* __restrict__ out,
                                              unsigned* __restrict__ bar){
  __shared__ __bf16 wlds[4*8*512];        // 32 KB: [gate][kt][32 lanes x 16]
  __shared__ __bf16 hlds[BATCH*HID_];     // 32 KB: h[t] stage (row-major)
  __shared__ float  gbuf[4][BATCH][16];   // 16 KB pre-activation gates
  __shared__ float  cbuf[BATCH][16];      //  4 KB cell state slice

  const int tid  = threadIdx.x;
  const int wg   = blockIdx.x;            // 0..15
  const int wave = tid >> 5;
  const int lane = tid & 31;
  const int half = lane >> 4;
  const int nlo  = lane & 15;
  const int g    = wave >> 2;             // gate 0..3
  const int mt   = wave & 3;              // batch tile 0..3

  // stage this WG's W_h fragments into LDS (4 ntiles x 8 ktiles x 1KB)
  for (int c = tid; c < 4*8*64; c += blockDim.x){   // chunks of 8 bf16 (16B)
    int lt  = c >> 6;                     // 0..31 local tile
    int off = (c & 63) * 8;
    int gg  = lt >> 3;
    int kt  = lt & 7;
    const __bf16* src = wfh + (size_t)(kt*64 + gg*16 + wg)*512 + off;
    *(bf16x8*)(wlds + lt*512 + off) = *(const bf16x8*)src;
  }
  for (int e = tid; e < BATCH*16; e += blockDim.x) ((float*)cbuf)[e] = 0.0f;
  __syncthreads();

  const int cP      = g*256 + wg*16 + nlo;    // column in P
  const int rowbase = mt*16 + half*8;
  const int arow    = mt*16 + nlo;

  for (int t = 0; t < SEQ; ++t){
    const __bf16* hsrc = hbuf + (size_t)(t & 1)*BATCH*HID_;
    __bf16*       hdst = hbuf + (size_t)((t+1) & 1)*BATCH*HID_;
    const float*  Pt   = P + (size_t)t*BATCH*NCOMB;
    if (t + 1 < SEQ)
      __builtin_prefetch(Pt + BATCH*NCOMB + (size_t)rowbase*NCOMB + cP, 0, 0);

    // ---- stage h[t] (32 KB) into LDS: async DMA if available ------------
#if HAVE_ASYNC_LDS
    for (int c = tid; c < (BATCH*HID_)/8; c += blockDim.x){       // 2048 x 16B
      __builtin_amdgcn_global_load_async_to_lds_b128(
          (AS1 v4i_gcc*)((AS1 const char*)hsrc + (size_t)c*16),
          (AS3 v4i_gcc*)((AS3 char*)hlds + (size_t)c*16), 0, 0);
    }
#else
    for (int c = tid; c < (BATCH*HID_)/8; c += blockDim.x)
      *(bf16x8*)(hlds + c*8) = *(const bf16x8*)(hsrc + c*8);
#endif

    // overlap: load P accumulator tile from global while the copy is in flight
    v8f acc;
#pragma unroll
    for (int r=0;r<8;++r) acc[r] = Pt[(size_t)(rowbase + r)*NCOMB + cP];

#if HAVE_ASYNC_LDS
#if __has_builtin(__builtin_amdgcn_s_wait_asynccnt)
    __builtin_amdgcn_s_wait_asynccnt(0);
#else
    asm volatile("s_wait_asynccnt 0x0" ::: "memory");
#endif
#endif
    __syncthreads();

    // ---- 16x16x256 WMMA chain, A from LDS (h), B from LDS (weights) -----
    const __bf16* Ar = hlds + (size_t)arow*HID_;
#pragma unroll
    for (int kt=0; kt<8; ++kt){
      int kc = kt*32 + half*16;
      bf16x8 a0 = *(const bf16x8*)(Ar + kc);
      bf16x8 a1 = *(const bf16x8*)(Ar + kc + 8);
      const __bf16* bp = wlds + (size_t)(g*8 + kt)*512 + lane*16;
      bf16x8 b0 = *(const bf16x8*)bp;
      bf16x8 b1 = *(const bf16x8*)(bp + 8);
      acc = __builtin_amdgcn_wmma_f32_16x16x32_bf16(false, make_frag(a0,a1),
                                                    false, make_frag(b0,b1),
                                                    (short)0, acc, false, false);
    }
#pragma unroll
    for (int r=0;r<8;++r) gbuf[g][rowbase + r][nlo] = acc[r];
    __syncthreads();

    // ---- elementwise cell update for this WG's 64x16 slice ---------------
    for (int e = tid; e < BATCH*16; e += blockDim.x){
      int row = e >> 4, jj = e & 15;
      float iv = sigmoidf_(gbuf[0][row][jj]);
      float fv = sigmoidf_(gbuf[1][row][jj]);
      float gv = tanhf_  (gbuf[2][row][jj]);
      float ov = sigmoidf_(gbuf[3][row][jj]);
      float cv = fv*cbuf[row][jj] + iv*gv;
      cbuf[row][jj] = cv;
      float hv = ov*tanhf_(cv);
      hdst[(size_t)row*HID_ + wg*16 + jj] = (__bf16)hv;
      if (t == SEQ-1){
        out[(size_t)row*HID_ + wg*16 + jj] = hv;                         // h_t
        out[(size_t)BATCH*HID_ + (size_t)row*HID_ + wg*16 + jj] = cv;    // c_t
      }
    }
    __syncthreads();

    // ---- device-wide sense-reversal barrier across the 16 persistent WGs -
    if (tid == 0){
      __threadfence();
      unsigned gen0 = __hip_atomic_load(bar+1, __ATOMIC_RELAXED, __HIP_MEMORY_SCOPE_AGENT);
      unsigned a = __hip_atomic_fetch_add(bar, 1u, __ATOMIC_ACQ_REL, __HIP_MEMORY_SCOPE_AGENT);
      if (a + 1u == (unsigned)NWG_LSTM){
        __hip_atomic_store(bar, 0u, __ATOMIC_RELAXED, __HIP_MEMORY_SCOPE_AGENT);
        __hip_atomic_store(bar+1, gen0+1u, __ATOMIC_RELEASE, __HIP_MEMORY_SCOPE_AGENT);
      } else {
        while (__hip_atomic_load(bar+1, __ATOMIC_ACQUIRE, __HIP_MEMORY_SCOPE_AGENT) == gen0)
          __builtin_amdgcn_s_sleep(1);
      }
      __threadfence();
    }
    __syncthreads();
  }
}

// ---------------------------------------------------------------------------
extern "C" void kernel_launch(void* const* d_in, const int* in_sizes, int n_in,
                              void* d_out, int out_size, void* d_ws, size_t ws_size,
                              hipStream_t stream){
  const float* x   = (const float*)d_in[0];
  const float* Wii = (const float*)d_in[1];
  const float* Whi = (const float*)d_in[2];
  const float* bii = (const float*)d_in[3];
  const float* bhi = (const float*)d_in[4];
  const float* Wif = (const float*)d_in[5];
  const float* Whf = (const float*)d_in[6];
  const float* bif = (const float*)d_in[7];
  const float* bhf = (const float*)d_in[8];
  const float* Wig = (const float*)d_in[9];
  const float* Whg = (const float*)d_in[10];
  const float* big = (const float*)d_in[11];
  const float* bhg = (const float*)d_in[12];
  const float* Wio = (const float*)d_in[13];
  const float* Who = (const float*)d_in[14];
  const float* bio = (const float*)d_in[15];
  const float* bho = (const float*)d_in[16];

  char* ws = (char*)d_ws;
  size_t off = 0;
  __bf16* xbf = (__bf16*)(ws + off); off += (size_t)MROWS*INF_*2;   // 64 MB
  __bf16* wfi = (__bf16*)(ws + off); off += (size_t)INF_*NCOMB*2;   // 512 KB
  __bf16* wfh = (__bf16*)(ws + off); off += (size_t)HID_*NCOMB*2;   // 512 KB
  float*  bias= (float*)(ws + off);  off += (size_t)NCOMB*4;        // 4 KB
  __bf16* hbuf= (__bf16*)(ws + off); off += (size_t)2*BATCH*HID_*2; // 64 KB (ping-pong)
  unsigned* bar=(unsigned*)(ws+off); off += 256;
  float*  P   = (float*)(ws + off);                                  // 512 MB [t][b][1024]
  float*  out = (float*)d_out;

  hipLaunchKernelGGL(k_init,  dim3(1),     dim3(256), 0, stream, hbuf, bar);
  hipLaunchKernelGGL(k_bias,  dim3(4),     dim3(256), 0, stream,
                     bii,bhi,bif,bhf,big,bhg,bio,bho, bias);
  hipLaunchKernelGGL(k_wprep, dim3(128),   dim3(256), 0, stream,
                     Wii,Whi,Wif,Whf,Wig,Whg,Wio,Who, wfi, wfh);
  hipLaunchKernelGGL(k_xconv, dim3((MROWS*INF_/4)/256), dim3(256), 0, stream, x, xbf);
  hipLaunchKernelGGL(k_gemm,  dim3((MROWS/16)*(NCOMB/16)/8), dim3(256), 0, stream,
                     xbf, wfi, bias, P);
  hipLaunchKernelGGL(k_lstm,  dim3(NWG_LSTM), dim3(512), 0, stream,
                     wfh, P, hbuf, out, bar);
}
